// MoEBlockANE_26525718020515
// MI455X (gfx1250) — matrix operand order repaired and checked by reading, not compiled, not run on hardware
//
#include <hip/hip_runtime.h>
#include <hip/hip_bf16.h>
#include <math.h>

#define D_DIM 640
#define I_DIM 640
#define N2    1280
#define E_NUM 32
#define T_NUM 64
#define K_TOP 4
#define EPS_V 1e-5f
#define LIMIT 7.0f

typedef __bf16        v16bf __attribute__((ext_vector_type(16)));
typedef float         v8f   __attribute__((ext_vector_type(8)));
typedef unsigned int  v4u   __attribute__((ext_vector_type(4)));

union BFVec {
    v16bf v;
    unsigned int d[8];
    unsigned short u[16];
    v4u q[2];
};

// round-to-nearest-even f32 -> bf16 (bit trick)
__device__ __forceinline__ unsigned short f2bf(float f) {
    unsigned int x = __float_as_uint(f);
    x += 0x7FFFu + ((x >> 16) & 1u);
    return (unsigned short)(x >> 16);
}
// pack two f32 -> one dword of two bf16 (lo, hi)
__device__ __forceinline__ unsigned int pack2bf(float lo, float hi) {
    return (unsigned int)f2bf(lo) | ((unsigned int)f2bf(hi) << 16);
}
// fast sigmoid: v_exp + v_rcp (avoids IEEE div chain)
__device__ __forceinline__ float fast_sigmoid(float z) {
    return __builtin_amdgcn_rcpf(1.0f + __expf(-z));
}

// ---------------------------------------------------------------------------
// Kernel 1: RMSNorm + router + top-4 + softmax + per-expert token lists +
//           bf16 activations + residual init of d_out. Tiny; single block.
// ---------------------------------------------------------------------------
__global__ __launch_bounds__(256) void moe_prep(
    const float* __restrict__ x, const float* __restrict__ norm_w,
    const float* __restrict__ gate_w, const float* __restrict__ gate_b,
    float* __restrict__ out, unsigned short* __restrict__ t_bf,
    int* __restrict__ cnt, int* __restrict__ tokid, float* __restrict__ tokw)
{
    __shared__ int   s_idx[T_NUM][K_TOP];
    __shared__ float s_w[T_NUM][K_TOP];
    const int tid = threadIdx.x;

    if (tid < T_NUM) {
        const int tok = tid;
        float ss = 0.f;
        for (int d = 0; d < D_DIM; ++d) {
            float v = x[d * T_NUM + tok];
            ss += v * v;
        }
        const float rinv = rsqrtf(ss * (1.0f / D_DIM) + EPS_V);

        float g[E_NUM];
#pragma unroll
        for (int e = 0; e < E_NUM; ++e) g[e] = gate_b[e];

        for (int d = 0; d < D_DIM; ++d) {
            float xn = x[d * T_NUM + tok] * rinv * norm_w[d];
            t_bf[tok * D_DIM + d] = f2bf(xn);
#pragma unroll
            for (int e = 0; e < E_NUM; ++e) g[e] += xn * gate_w[e * D_DIM + d];
        }

        // top-4, descending, lowest index wins ties (matches lax.top_k)
        unsigned mask = 0u;
        int   bi[K_TOP];
        float bv[K_TOP];
#pragma unroll
        for (int k = 0; k < K_TOP; ++k) {
            float best = -3.0e38f;
            int   besti = 0;
#pragma unroll
            for (int e = 0; e < E_NUM; ++e) {
                if ((mask >> e) & 1u) continue;
                if (g[e] > best) { best = g[e]; besti = e; }
            }
            mask |= 1u << besti;
            bi[k] = besti;
            bv[k] = best;
        }
        // softmax over the 4 (bv[0] is the max)
        float se = 0.f;
        float ew[K_TOP];
#pragma unroll
        for (int k = 0; k < K_TOP; ++k) { ew[k] = __expf(bv[k] - bv[0]); se += ew[k]; }
        const float inv = 1.0f / se;
#pragma unroll
        for (int k = 0; k < K_TOP; ++k) {
            s_idx[tok][k] = bi[k];
            s_w[tok][k]   = ew[k] * inv;
        }
    }
    __syncthreads();

    // deterministic per-expert list build: one thread per expert scans tokens
    if (tid < E_NUM) {
        const int e = tid;
        int c = 0;
        for (int tok = 0; tok < T_NUM; ++tok)
            for (int k = 0; k < K_TOP; ++k)
                if (s_idx[tok][k] == e) {
                    tokid[e * T_NUM + c] = tok;
                    tokw[e * T_NUM + c]  = s_w[tok][k];
                    ++c;
                }
        cnt[e] = c;
    }

    // residual init: out = x
    for (int i = tid; i < D_DIM * T_NUM; i += blockDim.x) out[i] = x[i];
}

// ---------------------------------------------------------------------------
// Kernel 2: h = t @ w1[e] + b1[e]; fused clamped-SiLU GLU -> hact (bf16).
// grid (E, 5); 8 waves/block; wave w owns one 16x16 glu tile + paired lin
// tile. K-loop is software-pipelined: loads for chunk i+1 are issued before
// converting/WMMA-ing chunk i (f32 staging regs), so VMEM overlaps WMMA.
// ---------------------------------------------------------------------------
__global__ __launch_bounds__(256) void moe_ffn1(
    const float* __restrict__ w1, const float* __restrict__ b1,
    const unsigned short* __restrict__ t_bf,
    const int* __restrict__ cnt, const int* __restrict__ tokid,
    unsigned short* __restrict__ hact)
{
    const int e = blockIdx.x;
    const int c = cnt[e];
    if (c == 0) return;

    const int wave = threadIdx.x >> 5;
    const int lane = threadIdx.x & 31;
    const int sel  = lane >> 4;   // lane half
    const int nn   = lane & 15;
    const int n0   = blockIdx.y * 128 + wave * 16;   // column base within I

    const float* W = w1 + (size_t)e * D_DIM * N2;
    const float bg1 = b1[e * N2 + n0 + nn];
    const float bl1 = b1[e * N2 + I_DIM + n0 + nn];
    const int colg = n0 + nn;
    const int coll = I_DIM + n0 + nn;

    const int mtiles = (c + 15) >> 4;
    for (int mt = 0; mt < mtiles; ++mt) {
        const int row  = mt * 16 + nn;
        const int rowc = row < c ? row : c - 1;     // clamp (EXEC must stay full)
        const int tokA = tokid[e * T_NUM + rowc];
        const v4u* Arow = (const v4u*)(t_bf + (size_t)tokA * D_DIM);

        // double-buffered staging: A as raw bf16 dwords, B as f32
        v4u   aq[2][2];
        float fg[2][16], fl[2][16];

        auto load_chunk = [&](int kb, int s) {
            const int ab = (kb >> 3) + sel;
            aq[s][0] = Arow[ab];
            aq[s][1] = Arow[ab + 2];
            const float* Wk = W + (size_t)(kb + sel * 8) * N2;
#pragma unroll
            for (int i = 0; i < 16; ++i) {
                fg[s][i] = Wk[(size_t)(i + (i & 8)) * N2 + colg];
                fl[s][i] = Wk[(size_t)(i + (i & 8)) * N2 + coll];
            }
        };

        v8f cg = {};
        v8f cl = {};

        auto do_mma = [&](int s) {
            BFVec a, bg, bl;
            a.q[0] = aq[s][0];
            a.q[1] = aq[s][1];
#pragma unroll
            for (int i = 0; i < 8; ++i) {
                bg.d[i] = pack2bf(fg[s][2 * i], fg[s][2 * i + 1]);
                bl.d[i] = pack2bf(fl[s][2 * i], fl[s][2 * i + 1]);
            }
            cg = __builtin_amdgcn_wmma_f32_16x16x32_bf16(false, a.v, false, bg.v,
                                                         (short)0, cg, false, false);
            cl = __builtin_amdgcn_wmma_f32_16x16x32_bf16(false, a.v, false, bl.v,
                                                         (short)0, cl, false, false);
        };

        load_chunk(0, 0);
#pragma unroll 1
        for (int kb = 0; kb < D_DIM; kb += 64) {
            load_chunk(kb + 32, 1);
            do_mma(0);
            if (kb + 64 < D_DIM) load_chunk(kb + 64, 0);
            do_mma(1);
        }

#pragma unroll
        for (int j = 0; j < 8; ++j) {
            const int m = mt * 16 + j + 8 * sel;    // C/D row layout
            if (m >= c) continue;
            float g = fminf(cg[j] + bg1, LIMIT);
            float l = fminf(fmaxf(cl[j] + bl1, -LIMIT), LIMIT);
            float act = g * fast_sigmoid(1.702f * g) * (l + 1.0f);
            hact[(size_t)(e * T_NUM + m) * I_DIM + n0 + nn] = f2bf(act);
        }
    }
}

// ---------------------------------------------------------------------------
// Kernel 3: out += ew * (hact @ w2[e] + b2[e]); atomic f32 accumulation
// into d_out which already holds the residual. grid (E, 5). Same pipelined
// K-loop structure as kernel 2.
// ---------------------------------------------------------------------------
__global__ __launch_bounds__(256) void moe_ffn2(
    const float* __restrict__ w2, const float* __restrict__ b2,
    const unsigned short* __restrict__ hact,
    const int* __restrict__ cnt, const int* __restrict__ tokid,
    const float* __restrict__ tokw, float* __restrict__ out)
{
    const int e = blockIdx.x;
    const int c = cnt[e];
    if (c == 0) return;

    const int wave = threadIdx.x >> 5;
    const int lane = threadIdx.x & 31;
    const int sel  = lane >> 4;
    const int nn   = lane & 15;
    const int col  = blockIdx.y * 128 + wave * 16 + nn;  // output d column

    const float* W = w2 + (size_t)e * I_DIM * D_DIM;
    const float bias = b2[e * D_DIM + col];

    const int mtiles = (c + 15) >> 4;
    for (int mt = 0; mt < mtiles; ++mt) {
        const int row  = mt * 16 + nn;
        const int rowc = row < c ? row : c - 1;
        const v4u* Arow = (const v4u*)(hact + (size_t)(e * T_NUM + rowc) * I_DIM);

        v4u   aq[2][2];
        float fb[2][16];

        auto load_chunk = [&](int kb, int s) {
            const int ab = (kb >> 3) + sel;
            aq[s][0] = Arow[ab];
            aq[s][1] = Arow[ab + 2];
            const float* Wk = W + (size_t)(kb + sel * 8) * D_DIM;
#pragma unroll
            for (int i = 0; i < 16; ++i)
                fb[s][i] = Wk[(size_t)(i + (i & 8)) * D_DIM + col];
        };

        v8f acc = {};

        auto do_mma = [&](int s) {
            BFVec a, b;
            a.q[0] = aq[s][0];
            a.q[1] = aq[s][1];
#pragma unroll
            for (int i = 0; i < 8; ++i)
                b.d[i] = pack2bf(fb[s][2 * i], fb[s][2 * i + 1]);
            acc = __builtin_amdgcn_wmma_f32_16x16x32_bf16(false, a.v, false, b.v,
                                                          (short)0, acc, false, false);
        };

        load_chunk(0, 0);
#pragma unroll 1
        for (int kb = 0; kb < I_DIM; kb += 64) {
            load_chunk(kb + 32, 1);
            do_mma(0);
            if (kb + 64 < I_DIM) load_chunk(kb + 64, 0);
            do_mma(1);
        }

#pragma unroll
        for (int j = 0; j < 8; ++j) {
            const int m = mt * 16 + j + 8 * sel;
            if (m >= c) continue;
            const int   tok = tokid[e * T_NUM + m];
            const float w   = tokw[e * T_NUM + m];
            atomicAdd(&out[(size_t)col * T_NUM + tok], (acc[j] + bias) * w);
        }
    }
}

// ---------------------------------------------------------------------------
extern "C" void kernel_launch(void* const* d_in, const int* in_sizes, int n_in,
                              void* d_out, int out_size, void* d_ws, size_t ws_size,
                              hipStream_t stream) {
    const float* x      = (const float*)d_in[0];
    const float* norm_w = (const float*)d_in[1];
    const float* gate_w = (const float*)d_in[2];
    const float* gate_b = (const float*)d_in[3];
    const float* w1     = (const float*)d_in[4];
    const float* b1     = (const float*)d_in[5];
    const float* w2     = (const float*)d_in[6];
    const float* b2     = (const float*)d_in[7];
    float* out = (float*)d_out;

    char* ws = (char*)d_ws;
    unsigned short* t_bf  = (unsigned short*)(ws + 0);        // 64*640*2  = 81920 B
    int*            cnt   = (int*)(ws + 81920);               // 32*4      =   128 B
    int*            tokid = (int*)(ws + 82048);               // 32*64*4   =  8192 B
    float*          tokw  = (float*)(ws + 90240);             // 32*64*4   =  8192 B
    unsigned short* hact  = (unsigned short*)(ws + 98432);    // 32*64*640*2 = 2.62 MB

    moe_prep<<<1, 256, 0, stream>>>(x, norm_w, gate_w, gate_b, out,
                                    t_bf, cnt, tokid, tokw);
    moe_ffn1<<<dim3(E_NUM, 5), 256, 0, stream>>>(w1, b1, t_bf, cnt, tokid, hact);
    moe_ffn2<<<dim3(E_NUM, 5), 256, 0, stream>>>(w2, b2, hact, cnt, tokid, tokw, out);
}
